// VisionEncoder_85358180041448
// MI455X (gfx1250) — matrix operand order repaired and checked by reading, not compiled, not run on hardware
//
#include <hip/hip_runtime.h>
#include <hip/hip_bf16.h>

typedef __attribute__((ext_vector_type(16))) _Float16 v16h;
typedef __attribute__((ext_vector_type(8)))  float    v8f;

#define BB     8
#define DD     768
#define NH     12
#define HDI    64
#define LREAL  577
#define LP     608     // padded seq len, multiple of 32
#define GG     24
#define DEPTHN 12
#define HIDN   3072

#define TM 128
#define TN 64

// ---------------- CDNA5 async global->LDS helpers ----------------

__device__ __forceinline__ void async_copy16(uint32_t lds_addr, const void* gaddr) {
  // GLOBAL_LOAD_ASYNC_TO_LDS_B128: per-lane 16B global -> LDS, tracked by ASYNCcnt
  asm volatile("global_load_async_to_lds_b128 %0, %1, off"
               :: "v"(lds_addr), "v"(gaddr)
               : "memory");
}

__device__ __forceinline__ void wait_async0() {
  asm volatile("s_wait_asynccnt 0" ::: "memory");
}

// ---------------- elementwise helpers ----------------

__global__ __launch_bounds__(256) void k_cast_f2h(const float* __restrict__ s,
                                                  _Float16* __restrict__ d, int n) {
  int i = blockIdx.x * 256 + threadIdx.x;
  if (i < n) d[i] = (_Float16)s[i];
}

__global__ __launch_bounds__(256) void k_rope(float* __restrict__ sinT,
                                              float* __restrict__ cosT) {
  int i = blockIdx.x * 256 + threadIdx.x;   // 576*64
  if (i >= 576 * 64) return;
  int pos = i >> 6, d = i & 63;
  int p = d >> 1;                           // comb repeated x2 along last dim
  int gy = pos / GG, gx = pos % GG;
  float coord; int f;
  if (p < 16) { coord = (float)gy; f = p; } else { coord = (float)gx; f = p - 16; }
  float invf = powf(10000.f, -(float)f * (1.f / 16.f));
  float a = coord * invf;
  sinT[i] = sinf(a);
  cosT[i] = cosf(a);
}

__global__ __launch_bounds__(256) void k_im2col(const float* __restrict__ x,
                                                _Float16* __restrict__ Pm, int n) {
  int i = blockIdx.x * 256 + threadIdx.x;
  if (i >= n) return;
  int col = i % 768;
  int patch = (i / 768) % 576;
  int b = i / (768 * 576);
  int c = col >> 8, rem = col & 255;
  int py = rem >> 4, px = rem & 15;
  int gy = patch / GG, gx = patch % GG;
  long long src = (((long long)(b * 3 + c) * 384) + gy * 16 + py) * 384 + gx * 16 + px;
  Pm[i] = (_Float16)x[src];
}

__global__ __launch_bounds__(256) void k_assemble(const float* __restrict__ ttmp,
                                                  const float* __restrict__ cls,
                                                  float* __restrict__ t, int n) {
  int i = blockIdx.x * 256 + threadIdx.x;
  if (i >= n) return;
  int d = i % DD;
  int l = (i / DD) % LP;
  int b = i / (DD * LP);
  float v = 0.f;
  if (l == 0) v = cls[d];
  else if (l <= 576) v = ttmp[((long long)(b * 576 + (l - 1))) * DD + d];
  t[i] = v;
}

__global__ __launch_bounds__(256) void k_concat(const float* __restrict__ O,
                                                _Float16* __restrict__ oh, int n) {
  int i = blockIdx.x * 256 + threadIdx.x;
  if (i >= n) return;
  int d = i % DD;
  int l = (i / DD) % LP;
  int b = i / (DD * LP);
  int h = d >> 6, dd = d & 63;
  oh[i] = (_Float16)O[(((long long)(b * NH + h)) * LP + l) * HDI + dd];
}

__global__ __launch_bounds__(256) void k_resadd(float* __restrict__ t,
                                                const float* __restrict__ y, int n) {
  int i = blockIdx.x * 256 + threadIdx.x;
  if (i >= n) return;
  int l = (i / DD) % LP;
  if (l < LREAL) t[i] += y[i];
}

__global__ __launch_bounds__(256) void k_gelu(const float* __restrict__ u,
                                              _Float16* __restrict__ g, int n) {
  int i = blockIdx.x * 256 + threadIdx.x;
  if (i >= n) return;
  float v = u[i];
  g[i] = (_Float16)(0.5f * v * (1.f + erff(v * 0.70710678118654752f)));
}

// ---------------- WMMA batched GEMM: C = A[M,K] * W[N,K]^T (+bias) ----------------
// 128x64 tile, 256 threads (8 waves in 4x2). Each wave computes a 32x32 region:
// 2x2 WMMA 16x16x32 subtiles (2 A-frags x 2 B-frags -> 4 WMMAs per K-step).
// Double-buffered LDS fed by async global->LDS (ASYNCcnt), one barrier per K-step.
// OOB rows are clamped (garbage only reaches discarded output rows/cols).

__global__ __launch_bounds__(256) void k_gemm(
    const _Float16* __restrict__ A, long long sAb,
    const _Float16* __restrict__ W, long long sWb,
    float* __restrict__ C, long long sCb,
    const float* __restrict__ bias,
    int M, int N, int K) {
  __shared__ __align__(32) _Float16 sA[2][TM * 32];
  __shared__ __align__(32) _Float16 sB[2][TN * 32];
  A += (long long)blockIdx.z * sAb;
  W += (long long)blockIdx.z * sWb;
  C += (long long)blockIdx.z * sCb;
  const int m0 = blockIdx.y * TM, n0 = blockIdx.x * TN;
  const int tid = threadIdx.x;
  const int wave = tid >> 5, lane = tid & 31;
  const int wm = (wave & 3) * 32;    // this wave's 32-row strip in the 128-row tile
  const int wn = (wave >> 2) * 32;   // this wave's 32-col strip in the 64-col tile

  // staging: 16B chunk per issue; row = c>>2, half-offset = (c&3)*8
  const int srow = tid >> 2;
  const int soff = (tid & 3) << 3;

  auto stage = [&](int p, int k0) {
    // A panel: 128 rows -> 2 chunks per thread
    int gm0 = m0 + srow;        if (gm0 >= M) gm0 = M - 1;
    int gm1 = m0 + srow + 64;   if (gm1 >= M) gm1 = M - 1;
    async_copy16((uint32_t)(uintptr_t)&sA[p][srow * 32 + soff],
                 A + (long long)gm0 * K + k0 + soff);
    async_copy16((uint32_t)(uintptr_t)&sA[p][(srow + 64) * 32 + soff],
                 A + (long long)gm1 * K + k0 + soff);
    // W panel: 64 rows -> 1 chunk per thread
    int gn = n0 + srow;         if (gn >= N) gn = N - 1;
    async_copy16((uint32_t)(uintptr_t)&sB[p][srow * 32 + soff],
                 W + (long long)gn * K + k0 + soff);
  };

  v8f acc[2][2];
  for (int i = 0; i < 2; ++i)
    for (int j = 0; j < 2; ++j)
      acc[i][j] = (v8f){0.f, 0.f, 0.f, 0.f, 0.f, 0.f, 0.f, 0.f};

  const int akb = (lane >> 4) << 3;   // A frag: K chunks {akb..+7, akb+16..+23}
  const int bkb = (lane >> 4) << 3;   // B frag: K chunks {bkb..+7, bkb+16..+23}
  const int al = lane & 15;

  int p = 0;
  stage(p, 0);
  wait_async0();
  __syncthreads();

  for (int k0 = 0; k0 < K; k0 += 32) {
    if (k0 + 32 < K) stage(p ^ 1, k0 + 32);

    union F { v16h v; uint4 u[2]; };
    F a[2], b[2];
#pragma unroll
    for (int si = 0; si < 2; ++si) {
      const _Float16* rp = &sA[p][(wm + si * 16 + al) * 32];
      a[si].u[0] = *(const uint4*)(rp + akb);
      a[si].u[1] = *(const uint4*)(rp + akb + 16);
    }
#pragma unroll
    for (int sj = 0; sj < 2; ++sj) {
      const _Float16* rp = &sB[p][(wn + sj * 16 + al) * 32];
      b[sj].u[0] = *(const uint4*)(rp + bkb);
      b[sj].u[1] = *(const uint4*)(rp + bkb + 16);
    }
#pragma unroll
    for (int si = 0; si < 2; ++si)
#pragma unroll
      for (int sj = 0; sj < 2; ++sj)
        acc[si][sj] = __builtin_amdgcn_wmma_f32_16x16x32_f16(
            false, a[si].v, false, b[sj].v, (short)0, acc[si][sj], false, false);

    wait_async0();
    __syncthreads();
    p ^= 1;
  }

#pragma unroll
  for (int si = 0; si < 2; ++si) {
#pragma unroll
    for (int sj = 0; sj < 2; ++sj) {
      int mbase = m0 + wm + si * 16 + (lane >> 4) * 8;
      int nc = n0 + wn + sj * 16 + al;
      if (nc < N) {
        float bv = bias ? bias[nc] : 0.f;
#pragma unroll
        for (int r = 0; r < 8; ++r) {
          int mm = mbase + r;
          if (mm < M) C[(long long)mm * N + nc] = acc[si][sj][r] + bv;
        }
      }
    }
  }
}

// ---------------- LayerNorm (row per block) + f16 cast ----------------

__global__ __launch_bounds__(256) void k_ln_cast(const float* __restrict__ t,
                                                 const float* __restrict__ g,
                                                 const float* __restrict__ b,
                                                 _Float16* __restrict__ o) {
  const int row = blockIdx.x, tid = threadIdx.x;
  const int l = row % LP;
  const float* src = t + (long long)row * DD;
  _Float16* dst = o + (long long)row * DD;
  if (l >= LREAL) {
    for (int i = tid; i < DD; i += 256) dst[i] = (_Float16)0.f;
    return;
  }
  float v0 = src[tid], v1 = src[tid + 256], v2 = src[tid + 512];
  float s = v0 + v1 + v2;
  float s2 = v0 * v0 + v1 * v1 + v2 * v2;
  for (int off = 16; off > 0; off >>= 1) {
    s  += __shfl_down(s, off);
    s2 += __shfl_down(s2, off);
  }
  __shared__ float red[16];
  if ((tid & 31) == 0) { red[tid >> 5] = s; red[8 + (tid >> 5)] = s2; }
  __syncthreads();
  float S = 0.f, S2 = 0.f;
  for (int w = 0; w < 8; ++w) { S += red[w]; S2 += red[8 + w]; }
  float mean = S * (1.f / DD);
  float var = S2 * (1.f / DD) - mean * mean;
  float inv = rsqrtf(var + 1e-5f);
  dst[tid]       = (_Float16)((v0 - mean) * inv * g[tid]       + b[tid]);
  dst[tid + 256] = (_Float16)((v1 - mean) * inv * g[tid + 256] + b[tid + 256]);
  dst[tid + 512] = (_Float16)((v2 - mean) * inv * g[tid + 512] + b[tid + 512]);
}

__global__ __launch_bounds__(256) void k_final_ln(const float* __restrict__ t,
                                                  const float* __restrict__ g,
                                                  const float* __restrict__ b,
                                                  float* __restrict__ o) {
  const int bb = blockIdx.x, tid = threadIdx.x;
  const float* src = t + (long long)bb * LP * DD;   // CLS row (l==0)
  float v0 = src[tid], v1 = src[tid + 256], v2 = src[tid + 512];
  float s = v0 + v1 + v2;
  float s2 = v0 * v0 + v1 * v1 + v2 * v2;
  for (int off = 16; off > 0; off >>= 1) {
    s  += __shfl_down(s, off);
    s2 += __shfl_down(s2, off);
  }
  __shared__ float red[16];
  if ((tid & 31) == 0) { red[tid >> 5] = s; red[8 + (tid >> 5)] = s2; }
  __syncthreads();
  float S = 0.f, S2 = 0.f;
  for (int w = 0; w < 8; ++w) { S += red[w]; S2 += red[8 + w]; }
  float mean = S * (1.f / DD);
  float var = S2 * (1.f / DD) - mean * mean;
  float inv = rsqrtf(var + 1e-5f);
  o[bb * DD + tid]       = (v0 - mean) * inv * g[tid]       + b[tid];
  o[bb * DD + tid + 256] = (v1 - mean) * inv * g[tid + 256] + b[tid + 256];
  o[bb * DD + tid + 512] = (v2 - mean) * inv * g[tid + 512] + b[tid + 512];
}

// ---------------- QKV split + 2D-RoPE + scale; v stored transposed ----------------

__global__ __launch_bounds__(256) void k_qkv_prep(const float* __restrict__ qkv,
                                                  const float* __restrict__ sinT,
                                                  const float* __restrict__ cosT,
                                                  _Float16* __restrict__ qh,
                                                  _Float16* __restrict__ kh,
                                                  _Float16* __restrict__ vt) {
  int idx = blockIdx.x * 4 + (threadIdx.x >> 6);   // (b*NH+h)*LP + l
  int d = threadIdx.x & 63;
  int l = idx % LP;
  int bh = idx / LP;
  int b = bh / NH, h = bh % NH;
  float q = 0.f, k = 0.f, v = 0.f;
  if (l < LREAL) {
    long long rb = ((long long)b * LP + l) * (3 * DD);
    int col = h * HDI + d;
    q = qkv[rb + col];
    k = qkv[rb + DD + col];
    v = qkv[rb + 2 * DD + col];
    if (l > 0) {   // RoPE on patch tokens only
      int pos = l - 1;
      float sn = sinT[pos * 64 + d], cs = cosT[pos * 64 + d];
      int pd = (d < 32) ? d + 32 : d - 32;
      float qp = qkv[rb + h * HDI + pd];
      float kp = qkv[rb + DD + h * HDI + pd];
      float rq = (d < 32) ? -qp : qp;
      float rk = (d < 32) ? -kp : kp;
      q = q * cs + rq * sn;
      k = k * cs + rk * sn;
    }
    q *= 0.125f;   // HD^-0.5
  }
  long long o = ((long long)bh * LP + l) * HDI + d;
  qh[o] = (_Float16)q;
  kh[o] = (_Float16)k;
  vt[((long long)bh * HDI + d) * LP + l] = (_Float16)v;   // [bh, dim, token]
}

// ---------------- masked softmax over real tokens, f16 output ----------------

__global__ __launch_bounds__(256) void k_softmax(const float* __restrict__ S,
                                                 _Float16* __restrict__ P) {
  const long long row = blockIdx.x;   // (b*NH+h)*LP + l
  const int tid = threadIdx.x;
  const int l = (int)(row % LP);
  const float* src = S + row * LP;
  _Float16* dst = P + row * LP;
  if (l >= LREAL) {
    for (int n = tid; n < LP; n += 256) dst[n] = (_Float16)0.f;
    return;
  }
  float mx = -3.4e38f;
  for (int j = 0; j < 3; ++j) {
    int n = tid + j * 256;
    if (n < LREAL) mx = fmaxf(mx, src[n]);
  }
  for (int off = 16; off > 0; off >>= 1) mx = fmaxf(mx, __shfl_down(mx, off));
  __shared__ float red[8];
  if ((tid & 31) == 0) red[tid >> 5] = mx;
  __syncthreads();
  float M = -3.4e38f;
  for (int w = 0; w < 8; ++w) M = fmaxf(M, red[w]);

  float ev[3];
  float s = 0.f;
  for (int j = 0; j < 3; ++j) {
    int n = tid + j * 256;
    float e = 0.f;
    if (n < LREAL) e = __expf(src[n] - M);
    ev[j] = e;
    s += e;
  }
  for (int off = 16; off > 0; off >>= 1) s += __shfl_down(s, off);
  __syncthreads();
  if ((tid & 31) == 0) red[tid >> 5] = s;
  __syncthreads();
  float Sum = 0.f;
  for (int w = 0; w < 8; ++w) Sum += red[w];
  float inv = 1.f / Sum;
  for (int j = 0; j < 3; ++j) {
    int n = tid + j * 256;
    if (n < LP) dst[n] = (n < LREAL) ? (_Float16)(ev[j] * inv) : (_Float16)0.f;
  }
}

// ---------------- host orchestration ----------------

extern "C" void kernel_launch(void* const* d_in, const int* in_sizes, int n_in,
                              void* d_out, int out_size, void* d_ws, size_t ws_size,
                              hipStream_t stream) {
  (void)in_sizes; (void)n_in; (void)out_size; (void)ws_size;
  const float* x       = (const float*)d_in[0];
  const float* patch_w = (const float*)d_in[1];
  const float* patch_b = (const float*)d_in[2];
  const float* cls     = (const float*)d_in[3];
  const float* ln1_s   = (const float*)d_in[4];
  const float* ln1_b   = (const float*)d_in[5];
  const float* qkv_w   = (const float*)d_in[6];
  const float* proj_w  = (const float*)d_in[7];
  const float* proj_b  = (const float*)d_in[8];
  const float* ln2_s   = (const float*)d_in[9];
  const float* ln2_b   = (const float*)d_in[10];
  const float* mlp_w1  = (const float*)d_in[11];
  const float* mlp_b1  = (const float*)d_in[12];
  const float* mlp_w2  = (const float*)d_in[13];
  const float* mlp_b2  = (const float*)d_in[14];
  const float* norm_s  = (const float*)d_in[15];
  const float* norm_b  = (const float*)d_in[16];
  float* out = (float*)d_out;

  size_t off = 0;
  char* base = (char*)d_ws;
  auto alloc = [&](size_t bytes) -> void* {
    off = (off + 255) & ~(size_t)255;
    void* p = base + off;
    off += bytes;
    return p;
  };

  _Float16* pw16    = (_Float16*)alloc((size_t)DD * 768 * 2);
  _Float16* qkvw16  = (_Float16*)alloc((size_t)DEPTHN * 3 * DD * DD * 2);
  _Float16* projw16 = (_Float16*)alloc((size_t)DEPTHN * DD * DD * 2);
  _Float16* w1_16   = (_Float16*)alloc((size_t)DEPTHN * HIDN * DD * 2);
  _Float16* w2_16   = (_Float16*)alloc((size_t)DEPTHN * DD * HIDN * 2);
  float*    sinT    = (float*)alloc((size_t)576 * 64 * 4);
  float*    cosT    = (float*)alloc((size_t)576 * 64 * 4);
  _Float16* Pm      = (_Float16*)alloc((size_t)BB * 576 * 768 * 2);
  float*    t       = (float*)alloc((size_t)BB * LP * DD * 4);
  _Float16* h16     = (_Float16*)alloc((size_t)BB * LP * DD * 2);
  float*    big     = (float*)alloc((size_t)BB * LP * HIDN * 4);
  _Float16* qh      = (_Float16*)alloc((size_t)BB * NH * LP * HDI * 2);
  _Float16* kh      = (_Float16*)alloc((size_t)BB * NH * LP * HDI * 2);
  _Float16* vt      = (_Float16*)alloc((size_t)BB * NH * LP * HDI * 2);
  float*    Sc      = (float*)alloc((size_t)BB * NH * LP * LP * 4);
  _Float16* Pp      = (_Float16*)alloc((size_t)BB * NH * LP * LP * 2);
  float*    Oa      = (float*)alloc((size_t)BB * NH * LP * HDI * 4);
  _Float16* oh      = (_Float16*)alloc((size_t)BB * LP * DD * 2);
  float*    y       = (float*)alloc((size_t)BB * LP * DD * 4);
  _Float16* g16     = (_Float16*)alloc((size_t)BB * LP * HIDN * 2);

  auto ew = [](int n) { return dim3((unsigned)((n + 255) / 256)); };
  auto gt = [](int N_, int M_) {
    return dim3((unsigned)((N_ + TN - 1) / TN), (unsigned)((M_ + TM - 1) / TM), 1);
  };

  // weight casts (f32 -> f16), once per call
  k_cast_f2h<<<ew(DD * 768), 256, 0, stream>>>(patch_w, pw16, DD * 768);
  k_cast_f2h<<<ew(DEPTHN * 3 * DD * DD), 256, 0, stream>>>(qkv_w, qkvw16, DEPTHN * 3 * DD * DD);
  k_cast_f2h<<<ew(DEPTHN * DD * DD), 256, 0, stream>>>(proj_w, projw16, DEPTHN * DD * DD);
  k_cast_f2h<<<ew(DEPTHN * HIDN * DD), 256, 0, stream>>>(mlp_w1, w1_16, DEPTHN * HIDN * DD);
  k_cast_f2h<<<ew(DEPTHN * DD * HIDN), 256, 0, stream>>>(mlp_w2, w2_16, DEPTHN * DD * HIDN);
  k_rope<<<ew(576 * 64), 256, 0, stream>>>(sinT, cosT);

  // patch embed
  k_im2col<<<ew(BB * 576 * 768), 256, 0, stream>>>(x, Pm, BB * 576 * 768);
  k_gemm<<<gt(DD, BB * 576), 256, 0, stream>>>(Pm, 0, pw16, 0, big, 0, patch_b,
                                               BB * 576, DD, 768);
  k_assemble<<<ew(BB * LP * DD), 256, 0, stream>>>(big, cls, t, BB * LP * DD);

  const int Mrows = BB * LP;
  for (int i = 0; i < DEPTHN; ++i) {
    k_ln_cast<<<Mrows, 256, 0, stream>>>(t, ln1_s + i * DD, ln1_b + i * DD, h16);
    k_gemm<<<gt(3 * DD, Mrows), 256, 0, stream>>>(h16, 0, qkvw16 + (size_t)i * 3 * DD * DD, 0,
                                                  big, 0, nullptr, Mrows, 3 * DD, DD);
    k_qkv_prep<<<(BB * NH * LP) / 4, 256, 0, stream>>>(big, sinT, cosT, qh, kh, vt);
    {   // scores = q k^T  (96 batched GEMMs)
      dim3 g((LP + TN - 1) / TN, (LP + TM - 1) / TM, BB * NH);
      k_gemm<<<g, 256, 0, stream>>>(qh, (long long)LP * HDI, kh, (long long)LP * HDI,
                                    Sc, (long long)LP * LP, nullptr, LP, LP, HDI);
    }
    k_softmax<<<BB * NH * LP, 256, 0, stream>>>(Sc, Pp);
    {   // O = P v  (via v^T as W[N,K])
      dim3 g((HDI + TN - 1) / TN, (LP + TM - 1) / TM, BB * NH);
      k_gemm<<<g, 256, 0, stream>>>(Pp, (long long)LP * LP, vt, (long long)HDI * LP,
                                    Oa, (long long)LP * HDI, nullptr, LP, HDI, LP);
    }
    k_concat<<<ew(BB * LP * DD), 256, 0, stream>>>(Oa, oh, BB * LP * DD);
    k_gemm<<<gt(DD, Mrows), 256, 0, stream>>>(oh, 0, projw16 + (size_t)i * DD * DD, 0,
                                              y, 0, proj_b + i * DD, Mrows, DD, DD);
    k_resadd<<<ew(Mrows * DD), 256, 0, stream>>>(t, y, Mrows * DD);

    k_ln_cast<<<Mrows, 256, 0, stream>>>(t, ln2_s + i * DD, ln2_b + i * DD, h16);
    k_gemm<<<gt(HIDN, Mrows), 256, 0, stream>>>(h16, 0, w1_16 + (size_t)i * HIDN * DD, 0,
                                                big, 0, mlp_b1 + i * HIDN, Mrows, HIDN, DD);
    k_gelu<<<ew(Mrows * HIDN), 256, 0, stream>>>(big, g16, Mrows * HIDN);
    k_gemm<<<gt(DD, Mrows), 256, 0, stream>>>(g16, 0, w2_16 + (size_t)i * DD * HIDN, 0,
                                              y, 0, mlp_b2 + i * DD, Mrows, DD, HIDN);
    k_resadd<<<ew(Mrows * DD), 256, 0, stream>>>(t, y, Mrows * DD);
  }

  k_final_ln<<<BB, 256, 0, stream>>>(t, norm_s, norm_b, out);
}
